// LSTM_53412213293616
// MI455X (gfx1250) — matrix operand order repaired and checked by reading, not compiled, not run on hardware
//
#include <hip/hip_runtime.h>

// ---------------------------------------------------------------------------
// LSTM forward for MI455X (gfx1250, wave32, WMMA).
// B=64, T=512, I=512, H=1024.
//
// Persistent-kernel design:
//   grid = 64 blocks (one per 16-wide hidden tile), 128 threads = 4 wave32.
//   wave w owns batch tile m0 = 16*w; block owns hidden tile h0 = 16*blockIdx.
//   Each wave computes 4 gate tiles (i,f,g,o) of 16x16 with K = [x_t | h] =
//   1536, via v_wmma_f32_16x16x32_bf16 (fp32 accumulate).
//   Weight slice (4 gates x 16 cols x 1536 K, bf16 = 192KB) staged once into
//   LDS with global_load_async_to_lds_b128 (ASYNCcnt); c-state tile stays in
//   registers for all 512 steps; h state is double-buffered bf16 in workspace
//   with one device-wide barrier per step.
// ---------------------------------------------------------------------------

#define B_   64
#define T_   512
#define I_   512
#define H_   1024
#define KT   (I_ + H_)     // 1536
#define NBLK 64
#define NTHR 128

typedef __attribute__((ext_vector_type(16))) __bf16 v16bf;
typedef __attribute__((ext_vector_type(8)))  float  v8f;

union FragBF {            // 32 bytes: one WMMA bf16 A/B operand
  v16bf v;
  uint4 q[2];
};

static __device__ __forceinline__ unsigned short f2bf(float f) {
  unsigned u = __float_as_uint(f);
  u += 0x7FFFu + ((u >> 16) & 1u);      // round-to-nearest-even
  return (unsigned short)(u >> 16);
}

static __device__ __forceinline__ float fast_sigmoid(float x) {
  // 1 / (1 + 2^(-x*log2e)) using v_exp_f32 / v_rcp_f32 (trans ops co-execute
  // with WMMA on CDNA5).
  float e = __builtin_amdgcn_exp2f(x * -1.4426950408889634f);
  return __builtin_amdgcn_rcpf(1.0f + e);
}

#if __has_builtin(__builtin_amdgcn_tanhf)
static __device__ __forceinline__ float fast_tanh(float x) {
  return __builtin_amdgcn_tanhf(x);
}
#else
static __device__ __forceinline__ float fast_tanh(float x) {
  return 2.0f * fast_sigmoid(2.0f * x) - 1.0f;
}
#endif

// ------------------------------ prep kernels -------------------------------

__global__ void k_f32_to_bf16(const float* __restrict__ src,
                              unsigned short* __restrict__ dst, int n) {
  int i = blockIdx.x * blockDim.x + threadIdx.x;
  if (i < n) dst[i] = f2bf(src[i]);
}

// Combined weight matrix, bf16:  Wcat[row][k], row in [0,4H), k in [0,1536):
//   k < 512   -> W_ih[row][k]
//   k >= 512  -> W_hh[row][k-512]
__global__ void k_wcat_bf16(const float* __restrict__ Wih,
                            const float* __restrict__ Whh,
                            unsigned short* __restrict__ wcat) {
  int i = blockIdx.x * blockDim.x + threadIdx.x;
  if (i >= 4 * H_ * KT) return;
  int row = i / KT;
  int k   = i - row * KT;
  float w = (k < I_) ? Wih[row * I_ + k] : Whh[row * H_ + (k - I_)];
  wcat[i] = f2bf(w);
}

__global__ void k_bias_init(const float* __restrict__ b_ih,
                            const float* __restrict__ b_hh,
                            float* __restrict__ bias,
                            unsigned* __restrict__ cnt) {
  int i = blockIdx.x * blockDim.x + threadIdx.x;
  if (i < 4 * H_) bias[i] = b_ih[i] + b_hh[i];
  if (i == 0) *cnt = 0u;   // reset grid barrier each launch (graph-replay safe)
}

// --------------------------- persistent LSTM -------------------------------

__launch_bounds__(NTHR, 1)
__global__ void k_lstm_persistent(const unsigned short* __restrict__ xbf,
                                  const unsigned short* __restrict__ wcat, // bf16 [4H][KT]
                                  unsigned short* __restrict__ hbuf,       // 2*B*H
                                  const float* __restrict__ bias,
                                  const float* __restrict__ c0,
                                  float* __restrict__ out,                 // [B][T][H]
                                  float* __restrict__ hN,                  // [B][H]
                                  float* __restrict__ cN,                  // [B][H]
                                  unsigned* __restrict__ cnt) {
  extern __shared__ unsigned short smem[];   // [64 rows = 4 gates x 16 N][1536 K]

  const int tid  = threadIdx.x;
  const int lane = tid & 31;
  const int wave = tid >> 5;
  const int h0   = blockIdx.x * 16;    // hidden tile
  const int m0   = wave * 16;          // batch tile

  // ---- one-time async DMA of this block's weight slice into LDS ----
  // 64 rows x 1536 bf16 = 192KB, copied as 12288 x 16B async transfers.
  // LDS destination is given by the VGPR operand (byte offset within our allocation).
  for (int idx = tid; idx < 64 * (KT / 8); idx += NTHR) {
    int rl  = idx / (KT / 8);           // 0..63 = g*16 + n
    int c8  = idx - rl * (KT / 8);      // 16B chunk index within row
    int row = (rl >> 4) * H_ + h0 + (rl & 15);
    unsigned lds_off = (unsigned)((rl * KT + c8 * 8) * 2);
    const unsigned short* gsrc = wcat + (size_t)row * KT + c8 * 8;
    asm volatile("global_load_async_to_lds_b128 %0, %1, off"
                 :: "v"(lds_off), "v"(gsrc)
                 : "memory");
  }
  asm volatile("s_wait_asynccnt 0x0" ::: "memory");
  __syncthreads();

  // ---- lane -> fragment coordinates (CDNA5 WMMA layouts) ----
  const int rA = lane & 15;            // A-matrix row M
  const int hi = lane >> 4;
  const int kA = hi * 8;               // A chunks at +kA and +kA+16
  const int nB = lane & 15;            // B-matrix column N
  const int kB = hi * 8;               // B chunks at +kB and +kB+16
  const int hh = h0 + nB;              // global hidden index for C/D column

  const float bI = bias[0 * H_ + hh];
  const float bF = bias[1 * H_ + hh];
  const float bG = bias[2 * H_ + hh];
  const float bO = bias[3 * H_ + hh];

  // persistent cell-state tile in registers (C/D layout: M = j + 8*hi, N = nB)
  float c[8];
#pragma unroll
  for (int j = 0; j < 8; ++j) {
    int b = m0 + j + 8 * hi;
    c[j] = c0[b * H_ + hh];
  }

  const unsigned short* xrow  = xbf + (size_t)(m0 + rA) * T_ * I_ + kA;
  const unsigned short* bbase = smem + (size_t)nB * KT + kB;

#pragma unroll 1
  for (int t = 0; t < T_; ++t) {
    const unsigned short* hcur  = hbuf + (size_t)(t & 1) * (B_ * H_);
    unsigned short*       hnext = hbuf + (size_t)((t + 1) & 1) * (B_ * H_);

    v8f accI, accF, accG, accO;
#pragma unroll
    for (int j = 0; j < 8; ++j) { accI[j] = bI; accF[j] = bF; accG[j] = bG; accO[j] = bO; }

    // prefetch next timestep's x A-fragment stream (global_prefetch_b8)
    if (t + 1 < T_) {
      __builtin_prefetch(xrow + (size_t)(t + 1) * I_, 0, 3);
      __builtin_prefetch(xrow + (size_t)(t + 1) * I_ + 256, 0, 3);
    }

    // ---- K part 1: x_t contribution (K = 0..511) ----
    const unsigned short* aptr = xrow + (size_t)t * I_;
#pragma unroll 4
    for (int k = 0; k < I_; k += 32) {
      FragBF a;
      a.q[0] = *(const uint4*)(aptr + k);
      a.q[1] = *(const uint4*)(aptr + k + 16);
      FragBF w0, w1, w2, w3;
      const unsigned short* bp = bbase + k;
      w0.q[0] = *(const uint4*)(bp + 0 * 16 * KT);
      w0.q[1] = *(const uint4*)(bp + 0 * 16 * KT + 16);
      w1.q[0] = *(const uint4*)(bp + 1 * 16 * KT);
      w1.q[1] = *(const uint4*)(bp + 1 * 16 * KT + 16);
      w2.q[0] = *(const uint4*)(bp + 2 * 16 * KT);
      w2.q[1] = *(const uint4*)(bp + 2 * 16 * KT + 16);
      w3.q[0] = *(const uint4*)(bp + 3 * 16 * KT);
      w3.q[1] = *(const uint4*)(bp + 3 * 16 * KT + 16);
      accI = __builtin_amdgcn_wmma_f32_16x16x32_bf16(false, a.v, false, w0.v, (short)0, accI, false, false);
      accF = __builtin_amdgcn_wmma_f32_16x16x32_bf16(false, a.v, false, w1.v, (short)0, accF, false, false);
      accG = __builtin_amdgcn_wmma_f32_16x16x32_bf16(false, a.v, false, w2.v, (short)0, accG, false, false);
      accO = __builtin_amdgcn_wmma_f32_16x16x32_bf16(false, a.v, false, w3.v, (short)0, accO, false, false);
    }

    // ---- K part 2: h contribution (K = 512..1535) ----
    const unsigned short* hptr = hcur + (size_t)(m0 + rA) * H_ + kA;
#pragma unroll 4
    for (int k = 0; k < H_; k += 32) {
      FragBF a;
      a.q[0] = *(const uint4*)(hptr + k);
      a.q[1] = *(const uint4*)(hptr + k + 16);
      FragBF w0, w1, w2, w3;
      const unsigned short* bp = bbase + I_ + k;
      w0.q[0] = *(const uint4*)(bp + 0 * 16 * KT);
      w0.q[1] = *(const uint4*)(bp + 0 * 16 * KT + 16);
      w1.q[0] = *(const uint4*)(bp + 1 * 16 * KT);
      w1.q[1] = *(const uint4*)(bp + 1 * 16 * KT + 16);
      w2.q[0] = *(const uint4*)(bp + 2 * 16 * KT);
      w2.q[1] = *(const uint4*)(bp + 2 * 16 * KT + 16);
      w3.q[0] = *(const uint4*)(bp + 3 * 16 * KT);
      w3.q[1] = *(const uint4*)(bp + 3 * 16 * KT + 16);
      accI = __builtin_amdgcn_wmma_f32_16x16x32_bf16(false, a.v, false, w0.v, (short)0, accI, false, false);
      accF = __builtin_amdgcn_wmma_f32_16x16x32_bf16(false, a.v, false, w1.v, (short)0, accF, false, false);
      accG = __builtin_amdgcn_wmma_f32_16x16x32_bf16(false, a.v, false, w2.v, (short)0, accG, false, false);
      accO = __builtin_amdgcn_wmma_f32_16x16x32_bf16(false, a.v, false, w3.v, (short)0, accO, false, false);
    }

    // ---- gates, cell update, h write-out ----
#pragma unroll
    for (int j = 0; j < 8; ++j) {
      int b   = m0 + j + 8 * hi;
      float ig = fast_sigmoid(accI[j]);
      float fg = fast_sigmoid(accF[j]);
      float gg = fast_tanh(accG[j]);
      float og = fast_sigmoid(accO[j]);
      float cn = fg * c[j] + ig * gg;
      c[j] = cn;
      float hv = og * fast_tanh(cn);
      out[((size_t)b * T_ + t) * H_ + hh] = hv;
      hnext[b * H_ + hh] = f2bf(hv);
      if (t == T_ - 1) hN[b * H_ + hh] = hv;
    }

    // ---- device-wide barrier: publish h_{t+1} before anyone reads it ----
    __threadfence();
    __syncthreads();
    if (tid == 0) {
      __hip_atomic_fetch_add(cnt, 1u, __ATOMIC_RELEASE, __HIP_MEMORY_SCOPE_AGENT);
      const unsigned target = (unsigned)NBLK * (unsigned)(t + 1);
      while (__hip_atomic_load(cnt, __ATOMIC_ACQUIRE, __HIP_MEMORY_SCOPE_AGENT) < target) {
        __builtin_amdgcn_s_sleep(8);
      }
    }
    __syncthreads();
    __threadfence();
  }

  // ---- final cell state ----
#pragma unroll
  for (int j = 0; j < 8; ++j) {
    int b = m0 + j + 8 * hi;
    cN[b * H_ + hh] = c[j];
  }
}

// ------------------------------- launcher ----------------------------------

extern "C" void kernel_launch(void* const* d_in, const int* in_sizes, int n_in,
                              void* d_out, int out_size, void* d_ws, size_t ws_size,
                              hipStream_t stream) {
  (void)in_sizes; (void)n_in; (void)out_size; (void)ws_size;

  const float* x   = (const float*)d_in[0];   // [B,T,I]
  const float* h0  = (const float*)d_in[1];   // [1,B,H]
  const float* c0  = (const float*)d_in[2];   // [1,B,H]
  const float* Wih = (const float*)d_in[3];   // [4H,I]
  const float* Whh = (const float*)d_in[4];   // [4H,H]
  const float* bih = (const float*)d_in[5];   // [4H]
  const float* bhh = (const float*)d_in[6];   // [4H]
  float* out = (float*)d_out;

  // workspace layout (all >=256B-aligned)
  char* ws = (char*)d_ws;
  const size_t xbf_bytes  = (size_t)B_ * T_ * I_ * 2;        // 32 MB
  const size_t wcat_bytes = (size_t)4 * H_ * KT * 2;         // 12 MB
  const size_t hbuf_bytes = (size_t)2 * B_ * H_ * 2;         // 256 KB
  const size_t bias_bytes = (size_t)4 * H_ * 4;              // 16 KB
  unsigned short* xbf  = (unsigned short*)(ws);
  unsigned short* wcat = (unsigned short*)(ws + xbf_bytes);
  unsigned short* hbuf = (unsigned short*)(ws + xbf_bytes + wcat_bytes);
  float*          bias = (float*)(ws + xbf_bytes + wcat_bytes + hbuf_bytes);
  unsigned*       cnt  = (unsigned*)(ws + xbf_bytes + wcat_bytes + hbuf_bytes + bias_bytes);

  // 1) x -> bf16 (parallel, one-shot)
  const int nx = B_ * T_ * I_;
  k_f32_to_bf16<<<(nx + 255) / 256, 256, 0, stream>>>(x, xbf, nx);
  // 2) combined weights -> bf16
  const int nw = 4 * H_ * KT;
  k_wcat_bf16<<<(nw + 255) / 256, 256, 0, stream>>>(Wih, Whh, wcat);
  // 3) h0 -> bf16 into buffer 0
  k_f32_to_bf16<<<(B_ * H_ + 255) / 256, 256, 0, stream>>>(h0, hbuf, B_ * H_);
  // 4) combined bias + barrier-counter reset
  k_bias_init<<<(4 * H_ + 255) / 256, 256, 0, stream>>>(bih, bhh, bias, cnt);

  // 5) persistent LSTM: 64 blocks (one per hidden tile), 192 KB dynamic LDS
  float* hN = out + (size_t)B_ * T_ * H_;
  float* cN = hN + (size_t)B_ * H_;
  const size_t lds_bytes = (size_t)64 * KT * sizeof(unsigned short);  // 196608
  k_lstm_persistent<<<NBLK, NTHR, lds_bytes, stream>>>(
      xbf, wcat, hbuf, bias, c0, out, hN, cN, cnt);
}
